// Diffusion_15436112462451
// MI455X (gfx1250) — matrix operand order repaired and checked by reading, not compiled, not run on hardware
//
#include <hip/hip_runtime.h>
#include <hip/hip_bf16.h>
#include <math.h>

typedef __attribute__((ext_vector_type(16))) _Float16 v16h;
typedef __attribute__((ext_vector_type(8)))  _Float16 v8h;
typedef __attribute__((ext_vector_type(8)))  float    v8f;

#define T_STEPS 100
#define BATCHN  2048

// workspace layout (bytes)
#define WS_SCHED 0                         // 500 f32 (5 arrays x 100)
#define WS_TEMB  4096                      // 100*32 f16 = 6400 B
#define WS_W1T   16384                     // 256*128 f16 = 65536 B (K padded 112->128, bias folded at k=112)
#define WS_W2T   (16384 + 65536)           // 256*256 f16 = 131072 B
#define WS_W3T   (16384 + 65536 + 131072)  // 256*256 f16
#define WS_W4T   (16384 + 65536 + 131072 + 131072) // 16*256 f16 = 8192 B

// Branchless mish: tanh(softplus(x)) == (e^2 + 2e) / (e^2 + 2e + 2), e = exp(x).
// med3(x, x, 15) == x (median of x, x, 15); e is not clamped.
__device__ __forceinline__ float mish_fast(float x) {
  float e = __expf(__builtin_amdgcn_fmed3f(x, x, 15.f));
  float n = __builtin_fmaf(e, e, e + e);     // e^2 + 2e
  return x * n * __builtin_amdgcn_rcpf(n + 2.f);
}

__device__ __forceinline__ float clamp1(float v) {   // clip to [-1,1], one v_med3_f32
  return __builtin_amdgcn_fmed3f(v, -1.f, 1.f);
}

// accurate-ish mish for the one-time prep kernel (not in hot path)
__device__ __forceinline__ float mish_prep(float x) {
  float sp = (x > 20.f) ? x : log1pf(__expf(x));
  return x * tanhf(sp);
}

// ---------------------------------------------------------------------------
// Prep kernel: VP schedule (fp64, matches numpy), 100 time embeddings,
// fp16 transposed/padded weights (b1 folded into w1t column 112).
// One 256-thread workgroup, runs once.
// ---------------------------------------------------------------------------
__global__ __launch_bounds__(256) void prep_kernel(
    const float* __restrict__ tW1, const float* __restrict__ tb1,
    const float* __restrict__ tW2, const float* __restrict__ tb2,
    const float* __restrict__ W1,  const float* __restrict__ b1,
    const float* __restrict__ W2,  const float* __restrict__ W3,
    const float* __restrict__ W4,  void* ws)
{
  float*    sched  = (float*)ws;
  _Float16* temb16 = (_Float16*)((char*)ws + WS_TEMB);
  _Float16* w1t    = (_Float16*)((char*)ws + WS_W1T);
  _Float16* w2t    = (_Float16*)((char*)ws + WS_W2T);
  _Float16* w3t    = (_Float16*)((char*)ws + WS_W3T);
  _Float16* w4t    = (_Float16*)((char*)ws + WS_W4T);
  const int tid = threadIdx.x;

  if (tid == 0) {  // VP schedule, double precision like numpy
    const double T = 100.0, bmax = 10.0, bmin = 0.1;
    double ac = 1.0;
    for (int t = 1; t <= T_STEPS; ++t) {
      double alpha = exp(-bmin / T - 0.5 * (bmax - bmin) * (2.0 * t - 1.0) / (T * T));
      double beta  = 1.0 - alpha;
      double ac_prev = ac;
      ac *= alpha;
      int i = t - 1;
      sched[0 * 100 + i] = (float)sqrt(1.0 / ac);
      sched[1 * 100 + i] = (float)sqrt(1.0 / ac - 1.0);
      sched[2 * 100 + i] = (float)(beta * sqrt(ac_prev) / (1.0 - ac));
      sched[3 * 100 + i] = (float)((1.0 - ac_prev) * sqrt(alpha) / (1.0 - ac));
      double pv = beta * (1.0 - ac_prev) / (1.0 - ac);
      if (pv < 1e-20) pv = 1e-20;
      sched[4 * 100 + i] = (float)log(pv);
    }
  }
  __syncthreads();

  // time MLP for every timestep value i = 0..99 (t uniform over batch)
  __shared__ float h[256];
  for (int s = 0; s < T_STEPS; ++s) {
    float e[32];
    const float neg = -__logf(10000.f) / 15.f;
    #pragma unroll
    for (int j = 0; j < 16; ++j) {
      float ang = (float)s * __expf((float)j * neg);
      e[j]      = __sinf(ang);
      e[j + 16] = __cosf(ang);
    }
    float acc = tb1[tid];
    #pragma unroll
    for (int k = 0; k < 32; ++k) acc += e[k] * tW1[k * 256 + tid];
    h[tid] = mish_prep(acc);
    __syncthreads();
    if (tid < 32) {
      float o = tb2[tid];
      for (int u = 0; u < 256; ++u) o += h[u] * tW2[u * 32 + tid];
      temb16[s * 32 + tid] = (_Float16)o;
    }
    __syncthreads();
  }

  // transpose + fp16-convert weights: wt[n][k] = W[k][n].
  // W1: pad K 112->128; column 112 carries b1 (input provides constant 1.0).
  for (int idx = tid; idx < 256 * 128; idx += 256) {
    int n = idx >> 7, k = idx & 127;
    _Float16 v;
    if (k < 112)       v = (_Float16)W1[k * 256 + n];
    else if (k == 112) v = (_Float16)b1[n];
    else               v = (_Float16)0.f;
    w1t[n * 128 + k] = v;
  }
  for (int idx = tid; idx < 256 * 256; idx += 256) {
    int n = idx >> 8, k = idx & 255;
    w2t[n * 256 + k] = (_Float16)W2[k * 256 + n];
    w3t[n * 256 + k] = (_Float16)W3[k * 256 + n];
  }
  for (int idx = tid; idx < 16 * 256; idx += 256) {
    int n = idx >> 8, k = idx & 255;
    w4t[n * 256 + k] = (_Float16)W4[k * 16 + n];
  }
}

// ---------------------------------------------------------------------------
// One 16xN layer for this wave: out = act(in[16xK] @ Wt^T [+ b]).
// A frags preloaded to registers (K/32 x v16h); rolled N loop does 2 tiles
// per iteration -> two independent WMMA chains for XDL latency hiding.
// Fragment layouts used here (contiguous k, A and B alike):
//   A: row = lane&15; lanes<16 hold K k0..k0+15,
//      lanes>=16 hold k0+16..k0+31.
//   B (transposed weights): col = n0+(lane&15), 16 contiguous halves at
//      K offset (lane>=16 ? 16 : 0).
//   C/D: row = r + 8*(lane>=16), col = lane&15.
// ---------------------------------------------------------------------------
template <int K, int N, bool MISH, bool BIAS>
__device__ __forceinline__ void layer(const _Float16* __restrict__ inp, int inStride,
                                      const _Float16* __restrict__ wt,
                                      const float* __restrict__ bias,
                                      _Float16* __restrict__ outp, int outStride,
                                      int lane)
{
  constexpr int KF = K / 32;
  const int col   = lane & 15;
  const int rhi   = (lane & 16) ? 8 : 0;
  const int koffA = (lane & 16) ? 16 : 0;
  const int koffB = (lane & 16) ? 16 : 0;
  const _Float16* arow = inp + (lane & 15) * inStride + koffA;

  v16h afr[KF];
  #pragma unroll
  for (int kf = 0; kf < KF; ++kf) {
    v8h alo = *(const v8h*)(arow + kf * 32);
    v8h ahi = *(const v8h*)(arow + kf * 32 + 8);
    #pragma unroll
    for (int j = 0; j < 8; ++j) { afr[kf][j] = alo[j]; afr[kf][j + 8] = ahi[j]; }
  }

  #pragma unroll 1
  for (int n0 = 0; n0 < N; n0 += 32) {
    const _Float16* brow0 = wt + (n0 + col) * K + koffB;
    const _Float16* brow1 = brow0 + 16 * K;
    if (n0 + 32 < N) {
      __builtin_prefetch(brow0 + 32 * K, 0, 0);
      __builtin_prefetch(brow1 + 32 * K, 0, 0);
    }
    v8f acc0, acc1;
    if (BIAS) {
      float bv0 = bias[n0 + col];
      float bv1 = bias[n0 + 16 + col];
      #pragma unroll
      for (int r = 0; r < 8; ++r) { acc0[r] = bv0; acc1[r] = bv1; }
    } else {
      #pragma unroll
      for (int r = 0; r < 8; ++r) { acc0[r] = 0.f; acc1[r] = 0.f; }
    }
    #pragma unroll
    for (int kf = 0; kf < KF; ++kf) {
      v16h b0 = *(const v16h*)(brow0 + kf * 32);
      v16h b1 = *(const v16h*)(brow1 + kf * 32);
      acc0 = __builtin_amdgcn_wmma_f32_16x16x32_f16(false, afr[kf], false, b0,
                                                    (short)0, acc0, false, false);
      acc1 = __builtin_amdgcn_wmma_f32_16x16x32_f16(false, afr[kf], false, b1,
                                                    (short)0, acc1, false, false);
    }
    #pragma unroll
    for (int r = 0; r < 8; ++r) {
      float v0 = acc0[r], v1 = acc1[r];
      if (MISH) { v0 = mish_fast(v0); v1 = mish_fast(v1); }
      outp[(r + rhi) * outStride + (n0 + col)]      = (_Float16)v0;
      outp[(r + rhi) * outStride + (n0 + 16 + col)] = (_Float16)v1;
    }
  }
}

// final 256->16 layer, returns raw accumulator (eps prediction)
__device__ __forceinline__ v8f layer4(const _Float16* __restrict__ inp,
                                      const _Float16* __restrict__ wt,
                                      const float* __restrict__ bias, int lane)
{
  const int col   = lane & 15;
  const int koffA = (lane & 16) ? 16 : 0;
  const int koffB = (lane & 16) ? 16 : 0;
  const _Float16* arow = inp + (lane & 15) * 256 + koffA;
  const _Float16* brow = wt + col * 256 + koffB;
  float bv = bias[col];
  v8f acc;
  #pragma unroll
  for (int r = 0; r < 8; ++r) acc[r] = bv;
  #pragma unroll
  for (int k0 = 0; k0 < 256; k0 += 32) {
    v8h alo = *(const v8h*)(arow + k0);
    v8h ahi = *(const v8h*)(arow + k0 + 8);
    v16h a;
    #pragma unroll
    for (int j = 0; j < 8; ++j) { a[j] = alo[j]; a[j + 8] = ahi[j]; }
    v16h b = *(const v16h*)(brow + k0);
    acc = __builtin_amdgcn_wmma_f32_16x16x32_f16(false, a, false, b,
                                                 (short)0, acc, false, false);
  }
  return acc;
}

// ---------------------------------------------------------------------------
// Persistent kernel: 64 blocks x 2 waves; each wave owns 16 batch rows and
// runs all 100 diffusion steps locally (rows are independent -> no barriers).
// ---------------------------------------------------------------------------
__global__ __launch_bounds__(64) void diff_main(
    const float* __restrict__ state, const float* __restrict__ x_init,
    const float* __restrict__ noise,
    const float* __restrict__ b2, const float* __restrict__ b3,
    const float* __restrict__ b4,
    const void* __restrict__ ws, float* __restrict__ out)
{
  __shared__ _Float16 sA[2][16][128];   // concat input: x|temb|state|1|pad
  __shared__ _Float16 sB1[2][16][256];
  __shared__ _Float16 sB2[2][16][256];

  const int tid  = threadIdx.x;
  const int w    = tid >> 5;
  const int lane = tid & 31;
  const int col  = lane & 15;
  const int rhi  = (lane & 16) ? 8 : 0;
  const int rowBase = blockIdx.x * 32 + w * 16;

  const float*    sched  = (const float*)ws;
  const _Float16* temb16 = (const _Float16*)((const char*)ws + WS_TEMB);
  const _Float16* w1t    = (const _Float16*)((const char*)ws + WS_W1T);
  const _Float16* w2t    = (const _Float16*)((const char*)ws + WS_W2T);
  const _Float16* w3t    = (const _Float16*)((const char*)ws + WS_W3T);
  const _Float16* w4t    = (const _Float16*)((const char*)ws + WS_W4T);

  // x kept in registers in the exact WMMA C-layout of this wave's tile
  float xr_[8];
  #pragma unroll
  for (int r = 0; r < 8; ++r) {
    int row = r + rhi;
    float xv = x_init[(rowBase + row) * 16 + col];
    xr_[r] = xv;
    sA[w][row][col] = (_Float16)xv;
  }
  // state -> concat cols 48..111 ; col 112 = 1.0 (bias lane for folded b1);
  // cols 113..127 zero
  for (int r = 0; r < 16; ++r) {
    sA[w][r][48 + lane]      = (_Float16)state[(rowBase + r) * 64 + lane];
    sA[w][r][48 + 32 + lane] = (_Float16)state[(rowBase + r) * 64 + 32 + lane];
    if (lane < 16) sA[w][r][112 + lane] = (lane == 0) ? (_Float16)1.f : (_Float16)0.f;
  }

  #pragma unroll 1
  for (int i = T_STEPS - 1; i >= 0; --i) {
    // prefetch next step's contiguous noise tile (1KB) while we compute
    {
      int kn = (T_STEPS - 1) - i + 1;
      if (kn < T_STEPS && lane < 16)
        __builtin_prefetch(noise + ((size_t)kn * BATCHN + rowBase + lane) * 16, 0, 0);
    }
    // broadcast this step's time embedding into concat cols 16..47
    _Float16 tv = temb16[i * 32 + lane];
    #pragma unroll 4
    for (int r = 0; r < 16; ++r) sA[w][r][16 + lane] = tv;

    layer<128, 256, true, false>(&sA[w][0][0], 128, w1t, nullptr, &sB1[w][0][0], 256, lane);
    layer<256, 256, true, true >(&sB1[w][0][0], 256, w2t, b2,      &sB2[w][0][0], 256, lane);
    layer<256, 256, true, true >(&sB2[w][0][0], 256, w3t, b3,      &sB1[w][0][0], 256, lane);
    v8f eps = layer4(&sB1[w][0][0], w4t, b4, lane);

    const float sr   = sched[0 * 100 + i];
    const float srm1 = sched[1 * 100 + i];
    const float c1   = sched[2 * 100 + i];
    const float c2   = sched[3 * 100 + i];
    const float sig  = (i != 0) ? __expf(0.5f * sched[4 * 100 + i]) : 0.f;
    const int   k    = (T_STEPS - 1) - i;   // noise uses scan index, not i

    #pragma unroll
    for (int r = 0; r < 8; ++r) {
      int row = r + rhi;
      int gRow = rowBase + row;
      float x  = xr_[r];
      float e  = eps[r];
      float xrec = clamp1(__builtin_fmaf(sr, x, -srm1 * e));
      float mean = __builtin_fmaf(c1, xrec, c2 * x);
      float nz   = noise[((size_t)k * BATCHN + gRow) * 16 + col];
      float xn   = __builtin_fmaf(sig, nz, mean);
      xr_[r] = xn;
      sA[w][row][col] = (_Float16)xn;
    }
  }

  #pragma unroll
  for (int r = 0; r < 8; ++r) {
    int row = r + rhi;
    out[(rowBase + row) * 16 + col] = clamp1(xr_[r]);
  }
}

extern "C" void kernel_launch(void* const* d_in, const int* in_sizes, int n_in,
                              void* d_out, int out_size, void* d_ws, size_t ws_size,
                              hipStream_t stream) {
  const float* state  = (const float*)d_in[0];
  const float* x_init = (const float*)d_in[1];
  const float* noise  = (const float*)d_in[2];
  const float* tW1    = (const float*)d_in[3];
  const float* tb1    = (const float*)d_in[4];
  const float* tW2    = (const float*)d_in[5];
  const float* tb2    = (const float*)d_in[6];
  const float* W1     = (const float*)d_in[7];
  const float* b1     = (const float*)d_in[8];
  const float* W2     = (const float*)d_in[9];
  const float* b2     = (const float*)d_in[10];
  const float* W3     = (const float*)d_in[11];
  const float* b3     = (const float*)d_in[12];
  const float* W4     = (const float*)d_in[13];
  const float* b4     = (const float*)d_in[14];

  prep_kernel<<<1, 256, 0, stream>>>(tW1, tb1, tW2, tb2, W1, b1, W2, W3, W4, d_ws);
  diff_main<<<BATCHN / 32, 64, 0, stream>>>(state, x_init, noise,
                                            b2, b3, b4, d_ws, (float*)d_out);
}
